// ModulatedConv2d_ONNX_44616120270917
// MI455X (gfx1250) — compile-verified, matrix operand order checked by reading, not run on hardware
//
#include <hip/hip_runtime.h>
#include <hip/hip_bf16.h>

typedef __attribute__((ext_vector_type(16))) __bf16 v16bf;
typedef __attribute__((ext_vector_type(8)))  float  v8f;
typedef __attribute__((ext_vector_type(4)))  int    v4i;

#define AS1 __attribute__((address_space(1)))
#define AS3 __attribute__((address_space(3)))

#if defined(__has_builtin)
#if __has_builtin(__builtin_amdgcn_global_load_async_to_lds_b128)
#define HAS_ASYNC_LDS 1
#endif
#if __has_builtin(__builtin_amdgcn_s_wait_asynccnt)
#define HAS_WAIT_ASYNC 1
#endif
#endif

constexpr int Bn   = 8;
constexpr int Cin  = 512;
constexpr int H    = 64;
constexpr int Wd   = 64;
constexpr int HW   = H * Wd;
constexpr int COUT = 512;
constexpr int SD   = 512;

constexpr int XTW  = 66;            // padded width (1-px halo each side)
constexpr int XTH  = 66;
constexpr size_t XT_ELEMS = (size_t)Bn * XTH * XTW * Cin;

constexpr int MT   = 128;           // WG tile: couts
constexpr int NT   = 64;            // WG tile: pixels
constexpr int APAD = 40;            // LDS pitch (halves): conflict-free 16B frags
constexpr int BPAD = 40;
constexpr int NSTAGE = 9 * 16;      // 9 taps x 16 chunks of 32 channels (divisible by 3)

// ---------------- modulation: s[b][c] = style[b]·mod_w[c] + mod_b[c] -------
__global__ void mod_kernel(const float* __restrict__ style,
                           const float* __restrict__ mw,
                           const float* __restrict__ mb,
                           float* __restrict__ s) {
    int idx = blockIdx.x * blockDim.x + threadIdx.x;
    if (idx >= Bn * Cin) return;
    int b = idx >> 9;
    int c = idx & (Cin - 1);
    const float* st = style + (size_t)b * SD;
    const float* w  = mw    + (size_t)c * SD;
    float acc = mb[c];
#pragma unroll 8
    for (int k = 0; k < SD; ++k) acc += st[k] * w[k];
    s[idx] = acc;
}

// ---------------- zero-fill padded input buffer ----------------------------
__global__ void fill0_kernel(uint4* __restrict__ p, int n) {
    int i = blockIdx.x * blockDim.x + threadIdx.x;
    if (i < n) p[i] = make_uint4(0u, 0u, 0u, 0u);
}

// -- modulate + NCHW->padded channels-last bf16: xt[b][y+1][x+1][c] ---------
__global__ __launch_bounds__(256)
void xmod_kernel(const float* __restrict__ x, const float* __restrict__ s,
                 __bf16* __restrict__ xt) {
    int b    = blockIdx.y;
    int pix0 = blockIdx.x * 64;
    int t    = threadIdx.x;
    __shared__ __bf16 T[32][72];
    int pl = t & 63;
    int g  = t >> 6;
    int pw = t >> 2;
    int co = (t & 3) * 8;
    int py = (pix0 + pw) >> 6;
    int px = (pix0 + pw) & 63;
    size_t obase = (((size_t)b * XTH + py + 1) * XTW + px + 1) * Cin;
    for (int cc0 = 0; cc0 < Cin; cc0 += 32) {
#pragma unroll
        for (int i = 0; i < 8; ++i) {
            int cl = g * 8 + i;
            int c  = cc0 + cl;
            float v = x[((size_t)b * Cin + c) * HW + pix0 + pl] * s[b * Cin + c];
            T[cl][pl] = (__bf16)v;
        }
        __syncthreads();
        __align__(16) __bf16 tmp[8];
#pragma unroll
        for (int j = 0; j < 8; ++j) tmp[j] = T[co + j][pw];
        *(uint4*)&xt[obase + cc0 + co] = *(const uint4*)tmp;
        __syncthreads();
    }
}

// ----- weight OIHW f32 -> [cout][tap][c] bf16 (A-address linear in stage) --
__global__ void wcvt_kernel(const float* __restrict__ w, __bf16* __restrict__ wt) {
    int o = blockIdx.x * blockDim.x + threadIdx.x;
    if (o >= 9 * COUT * Cin) return;
    int cout = o / (9 * Cin);
    int rem  = o % (9 * Cin);
    int tap  = rem >> 9;
    int c    = rem & (Cin - 1);
    wt[o] = (__bf16)w[((size_t)cout * Cin + c) * 9 + tap];
}

__device__ __forceinline__ v16bf ld16(const __bf16* p) {
    union { uint4 u[2]; v16bf v; } u;
    u.u[0] = *(const uint4*)(p);
    u.u[1] = *(const uint4*)(p + 8);
    return u.v;
}

__device__ __forceinline__ void wait_async0() {
#if defined(HAS_WAIT_ASYNC)
    __builtin_amdgcn_s_wait_asynccnt(0);
#else
    asm volatile("s_wait_asynccnt 0x0" ::: "memory");
#endif
}
__device__ __forceinline__ void wait_async3() {
#if defined(HAS_WAIT_ASYNC)
    __builtin_amdgcn_s_wait_asynccnt(3);
#else
    asm volatile("s_wait_asynccnt 0x3" ::: "memory");
#endif
}

// --------------------- main implicit-GEMM WMMA conv ------------------------
__global__ __launch_bounds__(256)
void conv_wmma_kernel(const __bf16* __restrict__ xt,
                      const __bf16* __restrict__ wt,
                      float* __restrict__ out) {
    __shared__ __align__(16) __bf16 As[3][MT * APAD];  // 3 x (128 couts x 32 k)
    __shared__ __align__(16) __bf16 Bs[3][NT * BPAD];  // 3 x (64 pixels x 32 k)

    int t     = threadIdx.x;
    int mbase = blockIdx.y * MT;
    int nbase = blockIdx.x * NT;
    int lane  = t & 31;
    int wave  = t >> 5;
    int wm    = wave & 3;
    int wn    = wave >> 2;
    int khalf = lane >> 4;
    int r16   = lane & 15;

    v8f acc[2][2] = {};

    // cooperative-load mappings
    int arow = t >> 1;
    int acol = (t & 1) * 16;
    int brow = t >> 2;
    int bcol = (t & 3) * 8;

    int p  = nbase + brow;
    int pb = p >> 12;
    int pp = p & (HW - 1);
    int py = pp >> 6;
    int px = pp & 63;

    // wt layout [cout][tap][c]: per-stage A offset is exactly 32 halves
    const __bf16* asrc_base = wt + (size_t)(mbase + arow) * (9 * Cin) + acol;
    const __bf16* bsrc_base =
        xt + (((size_t)pb * XTH + py) * XTW + px) * Cin + bcol;

    auto issue = [&](int buf, int st) {
        int tap = st >> 4;
        int kc  = (st & 15) << 5;
        int kh  = (tap * 11) >> 5;                   // tap / 3 for tap in [0,8]
        int kw  = tap - kh * 3;
        const __bf16* ag = asrc_base + (size_t)st * 32;
        const __bf16* bg = bsrc_base + (size_t)((kh * XTW + kw) << 9) + kc;
        __bf16* al = &As[buf][arow * APAD + acol];
        __bf16* bl = &Bs[buf][brow * BPAD + bcol];
#if defined(HAS_ASYNC_LDS)
        __builtin_amdgcn_global_load_async_to_lds_b128((AS1 v4i*)ag,
                                                       (AS3 v4i*)al, 0, 0);
        __builtin_amdgcn_global_load_async_to_lds_b128((AS1 v4i*)(ag + 8),
                                                       (AS3 v4i*)(al + 8), 0, 0);
        __builtin_amdgcn_global_load_async_to_lds_b128((AS1 v4i*)bg,
                                                       (AS3 v4i*)bl, 0, 0);
#else
        uint4 a0 = *(const uint4*)ag;
        uint4 a1 = *(const uint4*)(ag + 8);
        uint4 bv = *(const uint4*)bg;
        *(uint4*)al       = a0;
        *(uint4*)(al + 8) = a1;
        *(uint4*)bl       = bv;
#endif
    };

    auto compute = [&](int buf) {
        v16bf af[2], bfr[2];
#pragma unroll
        for (int sm = 0; sm < 2; ++sm)
            af[sm] = ld16(&As[buf][(wm * 32 + sm * 16 + r16) * APAD + khalf * 16]);
#pragma unroll
        for (int sn = 0; sn < 2; ++sn)
            bfr[sn] = ld16(&Bs[buf][(wn * 32 + sn * 16 + r16) * BPAD + khalf * 16]);
#pragma unroll
        for (int sm = 0; sm < 2; ++sm)
#pragma unroll
            for (int sn = 0; sn < 2; ++sn)
                acc[sm][sn] = __builtin_amdgcn_wmma_f32_16x16x32_bf16(
                    false, af[sm], false, bfr[sn],
                    (short)0, acc[sm][sn], false, false);
    };

    // one pipeline step: issue stage st+1 into bnxt, retire stage st, compute
    auto step = [&](int bcur, int bnxt, int st) {
        int st1 = st + 1;
        if (st1 < NSTAGE) {
            issue(bnxt, st1);
#if defined(HAS_ASYNC_LDS)
            wait_async3();       // retire stage st loads; st+1 stays in flight
#endif
        } else {
#if defined(HAS_ASYNC_LDS)
            wait_async0();
#endif
        }
        __syncthreads();
        compute(bcur);
    };

    // prologue: stage 0 into buffer 0
    issue(0, 0);

    for (int st = 0; st < NSTAGE; st += 3) {
        step(0, 1, st);
        step(1, 2, st + 1);
        step(2, 0, st + 2);
    }

    // epilogue: NCHW f32 output
#pragma unroll
    for (int sm = 0; sm < 2; ++sm) {
#pragma unroll
        for (int sn = 0; sn < 2; ++sn) {
            int coutb = mbase + wm * 32 + sm * 16;
            int pg    = nbase + wn * 32 + sn * 16 + r16;
            int ob    = pg >> 12;
            int opp   = pg & (HW - 1);
            float* op = out + (size_t)ob * COUT * HW + opp;
#pragma unroll
            for (int rr = 0; rr < 8; ++rr) {
                int m = coutb + rr + khalf * 8;
                op[(size_t)m * HW] = acc[sm][sn][rr];
            }
        }
    }
}

// --------------------------------- launch ----------------------------------
extern "C" void kernel_launch(void* const* d_in, const int* in_sizes, int n_in,
                              void* d_out, int out_size, void* d_ws, size_t ws_size,
                              hipStream_t stream) {
    const float* x     = (const float*)d_in[0];
    const float* style = (const float*)d_in[1];
    const float* wgt   = (const float*)d_in[2];
    const float* mod_w = (const float*)d_in[3];
    const float* mod_b = (const float*)d_in[4];
    float* out = (float*)d_out;

    // workspace layout
    char* ws = (char*)d_ws;
    float*  s  = (float*)ws;                                         // 16 KB
    __bf16* wt = (__bf16*)(ws + 16384);                              // 9*512*512 bf16
    __bf16* xt = (__bf16*)(ws + 16384 + (size_t)9 * COUT * Cin * 2); // padded input

    int xt_uint4 = (int)(XT_ELEMS * 2 / 16);
    fill0_kernel<<<(xt_uint4 + 255) / 256, 256, 0, stream>>>((uint4*)xt, xt_uint4);
    mod_kernel<<<(Bn * Cin + 255) / 256, 256, 0, stream>>>(style, mod_w, mod_b, s);
    xmod_kernel<<<dim3(HW / 64, Bn), 256, 0, stream>>>(x, s, xt);
    wcvt_kernel<<<(9 * COUT * Cin + 255) / 256, 256, 0, stream>>>(wgt, wt);
    conv_wmma_kernel<<<dim3(Bn * HW / NT, COUT / MT), 256, 0, stream>>>(xt, wt, out);
}
